// PentachoronCantorCompanion_46523085750375
// MI455X (gfx1250) — matrix operand order repaired and checked by reading, not compiled, not run on hardware
//
#include <hip/hip_runtime.h>
#include <hip/hip_bf16.h>

typedef unsigned int u32;
typedef unsigned short u16;
typedef unsigned long long u64;

typedef __attribute__((ext_vector_type(16))) __bf16 v16bf;
typedef __attribute__((ext_vector_type(8)))  float  v8f;

#define DIMX      512
#define HEADS     8
#define HEAD_DIM  64
#define SEQ       2048
#define BATCH     2
#define KNBR      128
#define NQKV      (3*DIMX)          // 1536
#define MROWS     (BATCH*SEQ)       // 4096
#define SCALE_F   (0.125f)          // 1/sqrt(64)

// ---------- bf16 helpers ----------
__device__ __forceinline__ u16 f2bf(float f) {
    u32 u = __float_as_uint(f);
    u32 r = u + 0x7fffu + ((u >> 16) & 1u);   // RNE
    return (u16)(r >> 16);
}
__device__ __forceinline__ float bflo(u32 packed) {  // low 16 bits -> f32
    return __uint_as_float(packed << 16);
}
__device__ __forceinline__ float bfhi(u32 packed) {  // high 16 bits -> f32
    return __uint_as_float(packed & 0xffff0000u);
}

union BV { u32 u[8]; v16bf v; };

// ---------- WMMA tile loaders (ISA 7.12.2 layouts, wave32) ----------
// A: 16x32 bf16 row-major, lda in elements. lane: hp=lane>>4, row=lane&15.
// VGPR j<4 : K = hp*8 + 2j,2j+1 ; VGPR j>=4 : K = 16 + hp*8 + 2(j-4),+1
__device__ __forceinline__ v16bf load_a_tile(const u16* A, int lda, int lane) {
    BV t;
    const int hp  = lane >> 4;
    const int row = lane & 15;
    const u16* base = A + (size_t)row * lda;
#pragma unroll
    for (int j = 0; j < 8; ++j) {
        int k0 = (j < 4) ? (hp * 8 + 2 * j) : (16 + hp * 8 + 2 * (j - 4));
        t.u[j] = *(const u32*)(base + k0);
    }
    return t.v;
}
// B: 32x16 bf16 (KxN). Source is W^T stored row-major (N,K), ldw in elements.
// lane: col = lane&15 (selects n row of W^T), hp = lane>>4.
// VGPR j<4 : K = hp*8 + 2j,2j+1 ; VGPR j>=4 : K = 16 + hp*8 + 2(j-4),+1
__device__ __forceinline__ v16bf load_b_tile(const u16* Wt, int ldw, int lane) {
    BV t;
    const int hp  = lane >> 4;
    const int col = lane & 15;
    const u16* base = Wt + (size_t)col * ldw;
#pragma unroll
    for (int j = 0; j < 8; ++j) {
        int k0 = (j < 4) ? (hp * 8 + 2 * j) : (16 + hp * 8 + 2 * (j - 4));
        t.u[j] = *(const u32*)(base + k0);
    }
    return t.v;
}

// ---------- K0a: x fp32 -> bf16 ----------
__global__ __launch_bounds__(256) void cvt_x_kernel(const float* __restrict__ x,
                                                    u16* __restrict__ xb, int n) {
    int i = blockIdx.x * 256 + threadIdx.x;
    if (i < n) xb[i] = f2bf(x[i]);
}

// ---------- K0b: W (K,N) fp32 -> W^T (N,K) bf16 ----------
__global__ __launch_bounds__(256) void transpose_w_kernel(const float* __restrict__ W,
                                                          u16* __restrict__ Wt,
                                                          int K, int N) {
    int i = blockIdx.x * 256 + threadIdx.x;
    if (i < K * N) {
        int n = i / K;
        int k = i - n * K;
        Wt[i] = f2bf(W[(size_t)k * N + n]);
    }
}

// ---------- K1a: sort coords ONCE (single block, LDS bitonic) ----------
// key = (float_bits(c) << 32) | idx ; coords are uniform [0,1) so the float
// bit pattern is monotonic. Produces sorted coords, sorted original indices,
// and rank-of-original-index.
__global__ __launch_bounds__(256) void sort_coords_kernel(const float* __restrict__ cc,
                                                          float* __restrict__ sc,
                                                          int* __restrict__ si,
                                                          int* __restrict__ pos) {
    __shared__ u64 keys[SEQ];
    for (int i = threadIdx.x; i < SEQ; i += 256) {
        keys[i] = ((u64)__float_as_uint(cc[i]) << 32) | (u32)i;
    }
    __syncthreads();
    for (int k = 2; k <= SEQ; k <<= 1) {
        for (int j = k >> 1; j > 0; j >>= 1) {
            for (int i = threadIdx.x; i < SEQ; i += 256) {
                int ixj = i ^ j;
                if (ixj > i) {
                    u64 a = keys[i], b = keys[ixj];
                    bool up = ((i & k) == 0);
                    if ((a > b) == up) { keys[i] = b; keys[ixj] = a; }
                }
            }
            __syncthreads();
        }
    }
    for (int i = threadIdx.x; i < SEQ; i += 256) {
        u64 kk = keys[i];
        int id = (int)(kk & 0xffffffffu);
        sc[i] = __uint_as_float((u32)(kk >> 32));
        si[i] = id;
        pos[id] = i;
    }
}

// ---------- K1b: 1-D kNN = contiguous window in sorted order ----------
// Two-pointer walk outward from the query's rank: 127 steps per query.
// Equal-distance ties broken toward the smaller original index (jax top_k).
__global__ __launch_bounds__(256) void window_routes_kernel(const float* __restrict__ sc_g,
                                                            const int* __restrict__ si_g,
                                                            const int* __restrict__ pos,
                                                            int* __restrict__ routes) {
    __shared__ float sc[SEQ];
    __shared__ int   si[SEQ];
    for (int i = threadIdx.x; i < SEQ; i += 256) { sc[i] = sc_g[i]; si[i] = si_g[i]; }
    __syncthreads();

    const int q = blockIdx.x * 256 + threadIdx.x;   // one query per thread
    const int p = pos[q];
    const float cq = sc[p];
    int l = p, r = p;
    int* out = routes + (size_t)q * KNBR;
    out[0] = q;                                     // dist 0
    for (int t = 1; t < KNBR; ++t) {
        const bool hasL = (l > 0);
        const bool hasR = (r < SEQ - 1);
        const float dl = hasL ? (cq - sc[l - 1]) : 3.4e38f;
        const float dr = hasR ? (sc[r + 1] - cq) : 3.4e38f;
        bool takeLeft;
        if (dl < dr)      takeLeft = true;
        else if (dr < dl) takeLeft = false;
        else              takeLeft = hasL && (!hasR || si[l - 1] < si[r + 1]);
        if (takeLeft) { --l; out[t] = si[l]; }
        else          { ++r; out[t] = si[r]; }
    }
}

// ---------- K2: QKV GEMM (bf16 WMMA), scatter epilogue to (B,H,S,64) ----------
// one wave -> 16x64 output tile; register double-buffered k-loop.
// M=4096, N=1536, K=512.
__global__ __launch_bounds__(128) void qkv_gemm_kernel(const u16* __restrict__ Xb,
                                                       const u16* __restrict__ Wt,
                                                       const float* __restrict__ bias,
                                                       u16* __restrict__ qb,
                                                       u16* __restrict__ kb,
                                                       u16* __restrict__ vb) {
    const int lane = threadIdx.x & 31;
    const int wid  = blockIdx.x * 4 + (threadIdx.x >> 5);
    const int mt = wid / (NQKV / 64);          // 0..255
    const int nc = wid % (NQKV / 64);          // 0..23
    const int m0 = mt * 16;
    const int n0 = nc * 64;

    v8f acc0 = {}, acc1 = {}, acc2 = {}, acc3 = {};
    const u16* Abase = Xb + (size_t)m0 * DIMX;
    const u16* B0 = Wt + (size_t)(n0 +  0) * DIMX;
    const u16* B1 = Wt + (size_t)(n0 + 16) * DIMX;
    const u16* B2 = Wt + (size_t)(n0 + 32) * DIMX;
    const u16* B3 = Wt + (size_t)(n0 + 48) * DIMX;

    v16bf a  = load_a_tile(Abase, DIMX, lane);
    v16bf b0 = load_b_tile(B0, DIMX, lane);
    v16bf b1 = load_b_tile(B1, DIMX, lane);
    v16bf b2 = load_b_tile(B2, DIMX, lane);
    v16bf b3 = load_b_tile(B3, DIMX, lane);

#pragma unroll
    for (int k0 = 0; k0 < DIMX; k0 += 32) {
        const int kn = k0 + 32;
        v16bf an = a, bn0 = b0, bn1 = b1, bn2 = b2, bn3 = b3;
        if (kn < DIMX) {                        // compile-time per unrolled iter
            __builtin_prefetch(Abase + kn + 32, 0, 1);
            an  = load_a_tile(Abase + kn, DIMX, lane);
            bn0 = load_b_tile(B0 + kn, DIMX, lane);
            bn1 = load_b_tile(B1 + kn, DIMX, lane);
            bn2 = load_b_tile(B2 + kn, DIMX, lane);
            bn3 = load_b_tile(B3 + kn, DIMX, lane);
        }
        acc0 = __builtin_amdgcn_wmma_f32_16x16x32_bf16(false, a, false, b0, (short)0, acc0, false, false);
        acc1 = __builtin_amdgcn_wmma_f32_16x16x32_bf16(false, a, false, b1, (short)0, acc1, false, false);
        acc2 = __builtin_amdgcn_wmma_f32_16x16x32_bf16(false, a, false, b2, (short)0, acc2, false, false);
        acc3 = __builtin_amdgcn_wmma_f32_16x16x32_bf16(false, a, false, b3, (short)0, acc3, false, false);
        a = an; b0 = bn0; b1 = bn1; b2 = bn2; b3 = bn3;
    }

    // epilogue: C VGPR r holds row m0 + r + 8*hp, col n0 + t*16 + (lane&15)
    const int hp   = lane >> 4;
    const int colL = lane & 15;
    v8f accs[4] = {acc0, acc1, acc2, acc3};
#pragma unroll
    for (int t = 0; t < 4; ++t) {
        const int nb    = n0 + t * 16;          // tile col base, multiple of 16
        const int which = nb >> 9;              // 0=q 1=k 2=v
        const int h     = (nb & 511) >> 6;
        const int d     = (nb & 63) + colL;
        u16* dst = (which == 0) ? qb : (which == 1) ? kb : vb;
        const float bv = bias[nb + colL];
#pragma unroll
        for (int r = 0; r < 8; ++r) {
            const int m  = m0 + r + 8 * hp;
            const int bI = m >> 11;
            const int s  = m & (SEQ - 1);
            dst[(((size_t)(bI * HEADS + h) * SEQ + s) * HEAD_DIM) + d] =
                f2bf(accs[t][r] + bv);
        }
    }
}

// ---------- K3: gathered softmax attention, one wave per (b,h,q) ----------
__global__ __launch_bounds__(256) void attn_kernel(const u16* __restrict__ qb,
                                                   const u16* __restrict__ kb,
                                                   const u16* __restrict__ vb,
                                                   const int* __restrict__ routes,
                                                   u16* __restrict__ attn_out) {
    __shared__ float qs[8][HEAD_DIM];
    const int lane = threadIdx.x & 31;
    const int w    = threadIdx.x >> 5;
    const int gq   = blockIdx.x * 8 + w;       // 0 .. B*H*S-1, flat (b,h,s)
    const int s    = gq & (SEQ - 1);
    const int bh   = gq >> 11;                 // b*8 + h

    // cache q in LDS (fp32)
    {
        u32 qq = *(const u32*)(qb + (size_t)gq * HEAD_DIM + 2 * lane);
        qs[w][2 * lane]     = bflo(qq);
        qs[w][2 * lane + 1] = bfhi(qq);
    }

    // scores: lane owns neighbor c*32+lane
    float sc[4];
    int   idx[4];
#pragma unroll
    for (int c = 0; c < 4; ++c) {
        const int n = c * 32 + lane;
        const int j = routes[s * KNBR + n];
        idx[c] = j;
        const u16* krow = kb + ((size_t)bh * SEQ + j) * HEAD_DIM;
        float dot = 0.f;
#pragma unroll
        for (int d = 0; d < HEAD_DIM; d += 8) {
            uint4 u = *(const uint4*)(krow + d);
            dot += qs[w][d + 0] * bflo(u.x) + qs[w][d + 1] * bfhi(u.x);
            dot += qs[w][d + 2] * bflo(u.y) + qs[w][d + 3] * bfhi(u.y);
            dot += qs[w][d + 4] * bflo(u.z) + qs[w][d + 5] * bfhi(u.z);
            dot += qs[w][d + 6] * bflo(u.w) + qs[w][d + 7] * bfhi(u.w);
        }
        sc[c] = dot * SCALE_F;
    }

    // softmax over 128 scores (4 per lane x 32 lanes)
    float m = fmaxf(fmaxf(sc[0], sc[1]), fmaxf(sc[2], sc[3]));
#pragma unroll
    for (int off = 16; off > 0; off >>= 1) m = fmaxf(m, __shfl_xor(m, off, 32));
    float p[4], sum = 0.f;
#pragma unroll
    for (int c = 0; c < 4; ++c) { p[c] = __expf(sc[c] - m); sum += p[c]; }
#pragma unroll
    for (int off = 16; off > 0; off >>= 1) sum += __shfl_xor(sum, off, 32);

    // output: lane owns dims (lane, lane+32)
    float acc0 = 0.f, acc1 = 0.f;
#pragma unroll 1
    for (int c = 0; c < 4; ++c) {
#pragma unroll
        for (int j = 0; j < 32; ++j) {
            const float pw = __shfl(p[c], j, 32);
            const int   id = __shfl(idx[c], j, 32);
            const u16* vrow = vb + ((size_t)bh * SEQ + id) * HEAD_DIM;
            acc0 += pw * bflo((u32)vrow[lane]);
            acc1 += pw * bflo((u32)vrow[lane + 32]);
        }
    }
    const float inv = 1.f / sum;
    acc0 *= inv; acc1 *= inv;

    const int bI = bh >> 3, h = bh & 7;
    u16* orow = attn_out + ((size_t)(bI * SEQ + s)) * DIMX + h * HEAD_DIM;
    orow[lane]      = f2bf(acc0);
    orow[lane + 32] = f2bf(acc1);
}

// ---------- K4: output projection GEMM (bf16 WMMA) + bias -> fp32 ----------
// M=4096, N=512, K=512; one wave -> 16x64 tile; double-buffered k-loop.
__global__ __launch_bounds__(128) void out_gemm_kernel(const u16* __restrict__ Ab,
                                                       const u16* __restrict__ Wt,
                                                       const float* __restrict__ bias,
                                                       float* __restrict__ out) {
    const int lane = threadIdx.x & 31;
    const int wid  = blockIdx.x * 4 + (threadIdx.x >> 5);
    const int mt = wid / (DIMX / 64);          // 0..255
    const int nc = wid % (DIMX / 64);          // 0..7
    const int m0 = mt * 16;
    const int n0 = nc * 64;

    v8f acc0 = {}, acc1 = {}, acc2 = {}, acc3 = {};
    const u16* Abase = Ab + (size_t)m0 * DIMX;
    const u16* B0 = Wt + (size_t)(n0 +  0) * DIMX;
    const u16* B1 = Wt + (size_t)(n0 + 16) * DIMX;
    const u16* B2 = Wt + (size_t)(n0 + 32) * DIMX;
    const u16* B3 = Wt + (size_t)(n0 + 48) * DIMX;

    v16bf a  = load_a_tile(Abase, DIMX, lane);
    v16bf b0 = load_b_tile(B0, DIMX, lane);
    v16bf b1 = load_b_tile(B1, DIMX, lane);
    v16bf b2 = load_b_tile(B2, DIMX, lane);
    v16bf b3 = load_b_tile(B3, DIMX, lane);

#pragma unroll
    for (int k0 = 0; k0 < DIMX; k0 += 32) {
        const int kn = k0 + 32;
        v16bf an = a, bn0 = b0, bn1 = b1, bn2 = b2, bn3 = b3;
        if (kn < DIMX) {
            __builtin_prefetch(Abase + kn + 32, 0, 1);
            an  = load_a_tile(Abase + kn, DIMX, lane);
            bn0 = load_b_tile(B0 + kn, DIMX, lane);
            bn1 = load_b_tile(B1 + kn, DIMX, lane);
            bn2 = load_b_tile(B2 + kn, DIMX, lane);
            bn3 = load_b_tile(B3 + kn, DIMX, lane);
        }
        acc0 = __builtin_amdgcn_wmma_f32_16x16x32_bf16(false, a, false, b0, (short)0, acc0, false, false);
        acc1 = __builtin_amdgcn_wmma_f32_16x16x32_bf16(false, a, false, b1, (short)0, acc1, false, false);
        acc2 = __builtin_amdgcn_wmma_f32_16x16x32_bf16(false, a, false, b2, (short)0, acc2, false, false);
        acc3 = __builtin_amdgcn_wmma_f32_16x16x32_bf16(false, a, false, b3, (short)0, acc3, false, false);
        a = an; b0 = bn0; b1 = bn1; b2 = bn2; b3 = bn3;
    }

    const int hp   = lane >> 4;
    const int colL = lane & 15;
    v8f accs[4] = {acc0, acc1, acc2, acc3};
#pragma unroll
    for (int t = 0; t < 4; ++t) {
        const int n  = n0 + t * 16 + colL;
        const float bv = bias[n];
#pragma unroll
        for (int r = 0; r < 8; ++r) {
            const int m = m0 + r + 8 * hp;
            out[(size_t)m * DIMX + n] = accs[t][r] + bv;
        }
    }
}

extern "C" void kernel_launch(void* const* d_in, const int* in_sizes, int n_in,
                              void* d_out, int out_size, void* d_ws, size_t ws_size,
                              hipStream_t stream) {
    const float* x    = (const float*)d_in[0];   // (2,2048,512)
    const float* cc   = (const float*)d_in[1];   // (2048,)
    const float* Wqkv = (const float*)d_in[2];   // (512,1536)
    const float* bqkv = (const float*)d_in[3];   // (1536,)
    const float* Wout = (const float*)d_in[4];   // (512,512)
    const float* bout = (const float*)d_in[5];   // (512,)
    float* out = (float*)d_out;                  // (2,2048,512)

    // workspace layout (bf16 elements unless noted)
    u16* xb     = (u16*)d_ws;                              // 2,097,152
    u16* wqkv_t = xb     + (size_t)MROWS * DIMX;           //   786,432
    u16* wout_t = wqkv_t + (size_t)NQKV * DIMX;            //   262,144
    u16* qb     = wout_t + (size_t)DIMX * DIMX;            // 2,097,152
    u16* kb     = qb     + (size_t)BATCH * HEADS * SEQ * HEAD_DIM;
    u16* vb     = kb     + (size_t)BATCH * HEADS * SEQ * HEAD_DIM;
    u16* attn   = vb     + (size_t)BATCH * HEADS * SEQ * HEAD_DIM;
    int* routes = (int*)(attn + (size_t)MROWS * DIMX);     // 262,144 ints
    float* sc_s = (float*)(routes + (size_t)SEQ * KNBR);   // 2048 floats
    int* si_s   = (int*)(sc_s + SEQ);                      // 2048 ints
    int* pos_s  = si_s + SEQ;                              // 2048 ints

    // 0) precision conversion / weight transpose
    {
        int n = MROWS * DIMX;
        cvt_x_kernel<<<(n + 255) / 256, 256, 0, stream>>>(x, xb, n);
    }
    {
        int n = DIMX * NQKV;
        transpose_w_kernel<<<(n + 255) / 256, 256, 0, stream>>>(Wqkv, wqkv_t, DIMX, NQKV);
    }
    {
        int n = DIMX * DIMX;
        transpose_w_kernel<<<(n + 255) / 256, 256, 0, stream>>>(Wout, wout_t, DIMX, DIMX);
    }

    // 1) neighbor routing: single sort + per-query two-pointer window
    sort_coords_kernel<<<1, 256, 0, stream>>>(cc, sc_s, si_s, pos_s);
    window_routes_kernel<<<SEQ / 256, 256, 0, stream>>>(sc_s, si_s, pos_s, routes);

    // 2) QKV projection on the matrix pipe
    {
        int waves = (MROWS / 16) * (NQKV / 64);   // 6144
        qkv_gemm_kernel<<<waves / 4, 128, 0, stream>>>(xb, wqkv_t, bqkv, qb, kb, vb);
    }

    // 3) gathered attention
    attn_kernel<<<(BATCH * HEADS * SEQ) / 8, 256, 0, stream>>>(qb, kb, vb, routes, attn);

    // 4) output projection on the matrix pipe (+bias, fp32 out)
    {
        int waves = (MROWS / 16) * (DIMX / 64);   // 2048
        out_gemm_kernel<<<waves / 4, 128, 0, stream>>>(attn, wout_t, bout, out);
    }
}
